// CellLayer_15264313770114
// MI455X (gfx1250) — matrix operand order repaired and checked by reading, not compiled, not run on hardware
//
#include <hip/hip_runtime.h>

// GRU: B=16, T=4096, D=256, H=512.
// 16 persistent workgroups (one per WGP), each owns 32 hidden units.
// Weights live in LDS as pre-packed bf16 WMMA B-fragments (144 KB).
// Hidden state is exchanged between WGPs as bf16 *A-fragments* in global
// memory and pulled in each step with the Tensor Data Mover (TDM), so the
// per-step LDS staging for the recurrent operand is a single 16 KB async DMA.

#define Bb 16
#define Tt 4096
#define Dd 256
#define Hh 512
#define NWG 16
#define NTHREADS 192              // 6 waves of 32
#define KSTEPS 24                 // K = 768 = 24 * 32
#define XKSTEPS 8                 // K 0..255  : x side
                                  // K 256..767: h side (ksteps 8..23, via TDM)

#define WELEMS (6 * KSTEPS * 32 * 16)    // weight fragments (bf16 elems)
#define AELEMS (KSTEPS * 32 * 16)        // activation fragments (bf16 elems)
#define WBYTES (WELEMS * 2)              // 147456
#define ABYTES (AELEMS * 2)              // 24576
#define GBYTES (8 * 16 * 16 * 4)         // 8192 (8 gate tiles, fp32)
#define HPBYTES (Bb * 32 * 4)            // 2048 (own fp32 h slice)
#define SMEM_BYTES (WBYTES + ABYTES + GBYTES + HPBYTES)

#define HFRAG_BYTES (16 * 32 * 16 * 2)   // 16 ksteps * 32 lanes * 16 bf16 = 16 KB
#define HFRAG_ELEMS8 (HFRAG_BYTES / 8)   // 2048 8-byte elements for TDM

typedef __attribute__((ext_vector_type(16))) __bf16 v16bf;
typedef __attribute__((ext_vector_type(8)))  float  v8f;
typedef __attribute__((ext_vector_type(4)))  unsigned int v4u;
typedef __attribute__((ext_vector_type(8)))  int    v8i;
typedef __attribute__((ext_vector_type(4)))  int    v4i;

__device__ __forceinline__ void grid_barrier(unsigned* cnt, unsigned* gen, unsigned nwg) {
  __syncthreads();
  if (threadIdx.x == 0) {
    __threadfence();  // publish h fragments device-wide before arriving
    unsigned g = __hip_atomic_load(gen, __ATOMIC_ACQUIRE, __HIP_MEMORY_SCOPE_AGENT);
    unsigned prev = __hip_atomic_fetch_add(cnt, 1u, __ATOMIC_ACQ_REL, __HIP_MEMORY_SCOPE_AGENT);
    if (prev == nwg - 1u) {
      __hip_atomic_store(cnt, 0u, __ATOMIC_RELAXED, __HIP_MEMORY_SCOPE_AGENT);
      __hip_atomic_fetch_add(gen, 1u, __ATOMIC_RELEASE, __HIP_MEMORY_SCOPE_AGENT);
    } else {
      while (__hip_atomic_load(gen, __ATOMIC_ACQUIRE, __HIP_MEMORY_SCOPE_AGENT) == g) {
        __builtin_amdgcn_s_sleep(2);
      }
    }
  }
  __syncthreads();
}

__device__ __forceinline__ float sigmoidf_(float x) {
  return 1.0f / (1.0f + __expf(-x));
}

__device__ __forceinline__ unsigned pack_bf16x2(float f0, float f1) {
  unsigned short u0 = __builtin_bit_cast(unsigned short, (__bf16)f0);
  unsigned short u1 = __builtin_bit_cast(unsigned short, (__bf16)f1);
  return (unsigned)u0 | ((unsigned)u1 << 16);
}

__global__ __launch_bounds__(NTHREADS, 1)
void gru_persistent(const float* __restrict__ xs,     // (B,T,D)
                    const float* __restrict__ Wih,    // (3H,D)
                    const float* __restrict__ Whh,    // (3H,H)
                    const float* __restrict__ bias,   // (3H)
                    const float* __restrict__ bn,     // (H)
                    float* __restrict__ out,          // (B,T,H)
                    unsigned* __restrict__ hfrag0,    // (16 KB) h A-fragments ping
                    unsigned* __restrict__ hfrag1,    // (16 KB) h A-fragments pong
                    unsigned* __restrict__ bar_cnt,
                    unsigned* __restrict__ bar_gen)
{
  extern __shared__ __align__(128) char smem[];
  __bf16* wfrag = (__bf16*)smem;                                 // [tile][ks][lane][16]
  __bf16* abuf  = (__bf16*)(smem + WBYTES);                      // [ks][lane][16]
  float*  gbuf  = (float*)(smem + WBYTES + ABYTES);              // [slot][16][16]
  float*  hprev = (float*)(smem + WBYTES + ABYTES + GBYTES);     // [16][32] own fp32 h

  const int wg    = blockIdx.x;        // 0..15 -> hidden cols [wg*32, wg*32+32)
  const int tid   = threadIdx.x;       // 0..191
  const int wave  = tid >> 5;          // 0..5 -> output tile
  const int lane  = tid & 31;
  const int jbase = wg * 32;

  // ---------------- setup: pack W = [W_ih | W_hh] into B-fragments ----------
  // B matrix is 32x16 (KxN): lane = khi*16 + n (khi = lane>>4),
  // dword v holds k_local = ((v>>2)*2 + khi)*8 + (v&3)*2 and +1.
  for (int idx = tid; idx < 6 * KSTEPS * 32 * 8; idx += NTHREADS) {
    int v    = idx & 7;
    int rest = idx >> 3;
    int ln   = rest & 31;
    int rest2 = rest >> 5;
    int ks   = rest2 % KSTEPS;
    int tile = rest2 / KSTEPS;
    int n    = ln & 15;
    int khi  = ln >> 4;
    int g    = tile >> 1, half = tile & 1;
    int R    = g * Hh + jbase + half * 16 + n;      // row of stacked W
    int k    = ks * 32 + (((v >> 2) << 1) + khi) * 8 + (v & 3) * 2;  // 0..767
    float f0, f1;
    if (k < Dd) { f0 = Wih[R * Dd + k];        f1 = Wih[R * Dd + k + 1]; }
    else        { f0 = Whh[R * Hh + (k - Dd)]; f1 = Whh[R * Hh + (k - Dd) + 1]; }
    *(unsigned*)&wfrag[((tile * KSTEPS + ks) * 32 + ln) * 16 + v * 2] = pack_bf16x2(f0, f1);
  }

  // own fp32 h slice starts at zero
  for (int idx = tid; idx < Bb * 32; idx += NTHREADS) hprev[idx] = 0.0f;

  // ---- TDM descriptor: 16 KB contiguous copy hfrag[t&1] -> abuf ksteps 8..23 ----
  // LDS offset = low 32 bits of the generic pointer (LDS aperture: addr[31:0]).
  const unsigned lds_hidden_off = (unsigned)(unsigned long long)(void*)(smem + WBYTES + XKSTEPS * 1024);
  unsigned long long ga0 = (unsigned long long)hfrag0;
  unsigned long long ga1 = (unsigned long long)hfrag1;
  v4u g0_even = { 1u, lds_hidden_off, (unsigned)(ga0 & 0xffffffffu),
                  (unsigned)((ga0 >> 32) & 0x1ffffffu) | (2u << 30) };
  v4u g0_odd  = { 1u, lds_hidden_off, (unsigned)(ga1 & 0xffffffffu),
                  (unsigned)((ga1 >> 32) & 0x1ffffffu) | (2u << 30) };
  // group1: data_size=8B, tensor_dim0=tile_dim0=stride0=2048, tensor_dim1=1
  v8i g1 = { (int)(3u << 16),
             (int)((HFRAG_ELEMS8 & 0xffffu) << 16),
             (int)((1u & 0xffffu) << 16),                 // tensor_dim1 = 1
             (int)((HFRAG_ELEMS8 & 0xffffu) << 16),       // tile_dim0
             (int)1,                                      // tile_dim1 = 1
             (int)HFRAG_ELEMS8,                           // tensor_dim0_stride
             0, 0 };
  v4i gz4 = { 0, 0, 0, 0 };
  v8i gz8 = { 0, 0, 0, 0, 0, 0, 0, 0 };

  // per-wave accumulator bias (same value down each column of the C tile)
  const int tile = wave;
  const int gsel = tile >> 1, halfsel = tile & 1;
  const int ncol = lane & 15;
  const int lhi  = lane >> 4;
  float bias_i, bias_h = 0.0f;
  if (tile < 4) {
    bias_i = bias[gsel * Hh + jbase + halfsel * 16 + ncol];
  } else {
    bias_i = bias[2 * Hh + jbase + halfsel * 16 + ncol];   // n-gate input side
    bias_h = bn[jbase + halfsel * 16 + ncol];              // b_n on hidden side
  }

  const __bf16* wbase = wfrag + tile * KSTEPS * 512 + lane * 16;
  const __bf16* abase = abuf + lane * 16;

  __syncthreads();

  // ---------------- time scan ----------------
  for (int t = 0; t < Tt; ++t) {
    unsigned* hfrag_next = (t & 1) ? hfrag0 : hfrag1;

    // (1) kick the TDM: pull full hidden-state fragments (16 KB) into LDS
    if (wave == 0) {
      __builtin_amdgcn_tensor_load_to_lds((t & 1) ? g0_odd : g0_even, g1, gz4, gz4, gz8, 0);
    }

    // (2) pack x_t into A-fragments for ksteps 0..7 (K = 0..255)
    //     A is 16x32 (MxK): lane m=ln&15; lanes 0-15 hold K groups {0-7,16-23},
    //     lanes 16-31 hold {8-15,24-31}.
    for (int idx = tid; idx < XKSTEPS * 32 * 8; idx += NTHREADS) {
      int v     = idx & 7;
      int ln    = (idx >> 3) & 31;
      int ks    = idx >> 8;
      int m     = ln & 15;
      int hi    = ln >> 4;
      int group = ((v >> 2) << 1) + hi;             // 0..3
      int k     = ks * 32 + group * 8 + (v & 3) * 2;
      const float2 xv = *(const float2*)&xs[(size_t)m * Tt * Dd + (size_t)t * Dd + k];
      *(unsigned*)&abuf[(ks * 32 + ln) * 16 + v * 2] = pack_bf16x2(xv.x, xv.y);
    }

    // (3) wait for the tensor DMA, then make LDS coherent across waves
    if (wave == 0) __builtin_amdgcn_s_wait_tensorcnt(0);
    __syncthreads();

    // (4) WMMA: D = A(16x768) x B(768x16) + bias, fp32 accumulate
    v8f acc  = {bias_i, bias_i, bias_i, bias_i, bias_i, bias_i, bias_i, bias_i};
    v8f acc2 = {bias_h, bias_h, bias_h, bias_h, bias_h, bias_h, bias_h, bias_h};
    if (tile < 4) {            // r / z gates: fully fused K = 768
#pragma unroll
      for (int ks = 0; ks < KSTEPS; ++ks) {
        v16bf av = *(const v16bf*)(abase + ks * 512);
        v16bf wv = *(const v16bf*)(wbase + ks * 512);
        acc = __builtin_amdgcn_wmma_f32_16x16x32_bf16(false, av, false, wv,
                                                      (short)0, acc, false, false);
      }
    } else {                   // n gate: input part and hidden part kept apart
#pragma unroll
      for (int ks = 0; ks < XKSTEPS; ++ks) {           // K 0..255 (x side)
        v16bf av = *(const v16bf*)(abase + ks * 512);
        v16bf wv = *(const v16bf*)(wbase + ks * 512);
        acc = __builtin_amdgcn_wmma_f32_16x16x32_bf16(false, av, false, wv,
                                                      (short)0, acc, false, false);
      }
#pragma unroll
      for (int ks = XKSTEPS; ks < KSTEPS; ++ks) {      // K 256..767 (h side)
        v16bf av = *(const v16bf*)(abase + ks * 512);
        v16bf wv = *(const v16bf*)(wbase + ks * 512);
        acc2 = __builtin_amdgcn_wmma_f32_16x16x32_bf16(false, av, false, wv,
                                                       (short)0, acc2, false, false);
      }
    }

    // (5) stage gate tiles: C layout row m = r + 8*(lane>>4), col = lane&15
    {
      float* gslot = gbuf + tile * 256;
#pragma unroll
      for (int r = 0; r < 8; ++r)
        gslot[(r + 8 * lhi) * 16 + ncol] = acc[r];
      if (tile >= 4) {
        float* gslot2 = gbuf + (tile + 2) * 256;
#pragma unroll
        for (int r = 0; r < 8; ++r)
          gslot2[(r + 8 * lhi) * 16 + ncol] = acc2[r];
      }
    }
    __syncthreads();

    // (6) elementwise GRU update, pairs (jj, jj+1); publish h_new as:
    //     fp32 -> out and local hprev; bf16 A-fragment dword -> hfrag_next.
    //     Our 32 columns land in kstep 8+wg => one contiguous 1 KB region.
    for (int idx = tid; idx < Bb * 16; idx += NTHREADS) {
      int m  = idx >> 4;
      int jj = (idx & 15) * 2;
      int half = jj >> 4, n = jj & 15;
      float hn0, hn1;
      {
        float rg  = gbuf[(0 + half) * 256 + m * 16 + n];
        float zg  = gbuf[(2 + half) * 256 + m * 16 + n];
        float nin = gbuf[(4 + half) * 256 + m * 16 + n];
        float nh  = gbuf[(6 + half) * 256 + m * 16 + n];
        float r = sigmoidf_(rg), z = sigmoidf_(zg);
        float nv = tanhf(nin + r * nh);
        float hp = hprev[m * 32 + jj];
        hn0 = nv + z * (hp - nv);
      }
      {
        float rg  = gbuf[(0 + half) * 256 + m * 16 + n + 1];
        float zg  = gbuf[(2 + half) * 256 + m * 16 + n + 1];
        float nin = gbuf[(4 + half) * 256 + m * 16 + n + 1];
        float nh  = gbuf[(6 + half) * 256 + m * 16 + n + 1];
        float r = sigmoidf_(rg), z = sigmoidf_(zg);
        float nv = tanhf(nin + r * nh);
        float hp = hprev[m * 32 + jj + 1];
        hn1 = nv + z * (hp - nv);
      }
      hprev[m * 32 + jj]     = hn0;
      hprev[m * 32 + jj + 1] = hn1;
      *(float2*)&out[(size_t)m * Tt * Hh + (size_t)t * Hh + jbase + jj] = make_float2(hn0, hn1);
      // A-fragment slot for k_local = jj: group=jj>>3, lane=m+16*(group&1),
      // v = (group>>1)*4 + ((jj&7)>>1); dword index within our 1 KB region:
      int group = jj >> 3;
      int lane_f = m + 16 * (group & 1);
      int v_f = (group >> 1) * 4 + ((jj & 7) >> 1);
      hfrag_next[wg * 256 + lane_f * 8 + v_f] = pack_bf16x2(hn0, hn1);
    }

    // prefetch next timestep's x rows while we wait at the barrier
    if (t + 1 < Tt && tid < Bb) {
      __builtin_prefetch(&xs[(size_t)tid * Tt * Dd + (size_t)(t + 1) * Dd], 0, 1);
    }

    grid_barrier(bar_cnt, bar_gen, (unsigned)gridDim.x);
  }
}

__global__ void gru_init(unsigned* __restrict__ hfrags, unsigned* __restrict__ ctrs) {
  int i = blockIdx.x * blockDim.x + threadIdx.x;
  if (i < (2 * HFRAG_BYTES) / 4) hfrags[i] = 0u;   // both h fragment buffers = bf16 zeros
  if (i < 2) ctrs[i] = 0u;
}

extern "C" void kernel_launch(void* const* d_in, const int* in_sizes, int n_in,
                              void* d_out, int out_size, void* d_ws, size_t ws_size,
                              hipStream_t stream) {
  (void)in_sizes; (void)n_in; (void)out_size; (void)ws_size;
  const float* xs  = (const float*)d_in[0];
  const float* Wih = (const float*)d_in[1];
  const float* Whh = (const float*)d_in[2];
  const float* b   = (const float*)d_in[3];
  const float* bn  = (const float*)d_in[4];
  float* out = (float*)d_out;

  char* ws = (char*)d_ws;
  unsigned* hfrag0 = (unsigned*)ws;                       // 16 KB
  unsigned* hfrag1 = (unsigned*)(ws + HFRAG_BYTES);       // 16 KB
  unsigned* ctrs   = (unsigned*)(ws + 2 * HFRAG_BYTES);

  gru_init<<<((2 * HFRAG_BYTES / 4) + 255) / 256, 256, 0, stream>>>(hfrag0, ctrs);
  gru_persistent<<<NWG, NTHREADS, SMEM_BYTES, stream>>>(
      xs, Wih, Whh, b, bn, out, hfrag0, hfrag1, ctrs, ctrs + 1);
}
